// ObjectDetector_58471684768044
// MI455X (gfx1250) — compile-verified, matrix operand order checked by reading, not run on hardware
//
#include <hip/hip_runtime.h>
#include <hip/hip_bf16.h>
#include <stdint.h>
#include <stddef.h>

// ---------------------------------------------------------------------------
// CDNA5 (gfx1250) object-detection head: implicit-GEMM conv via
// v_wmma_f32_16x16x32_bf16 with TDM (tensor_load_to_lds) double-buffered
// weight staging, fused BN+ReLU, anchor decode, top-K, greedy NMS.
// wave32 everywhere; 256-thread blocks = 8 waves for the GEMM kernels.
// ---------------------------------------------------------------------------

typedef __attribute__((ext_vector_type(16))) __bf16 v16bf;
typedef __attribute__((ext_vector_type(8)))  float  v8f;
typedef __attribute__((ext_vector_type(4)))  unsigned int u32x4;
typedef __attribute__((ext_vector_type(8)))  int          i32x8;
typedef __attribute__((ext_vector_type(4)))  int          i32x4;

#define GRID_HW   20
#define HW        400            // 20*20
#define BATCH     64
#define CIN       256
#define KTOT      2304           // 9 * 256
#define KC_STEPS  72             // KTOT / 32
#define MTOT      (BATCH * HW)   // 25600
#define NANCH     3600           // HW * 9
#define KCAND     300
#define MAXDET    100

__device__ __forceinline__ unsigned short f2bf(float f) {
    union { float f; unsigned u; } c; c.f = f;
    unsigned u = c.u;
    unsigned r = u + 0x7fffu + ((u >> 16) & 1u);   // round-to-nearest-even
    return (unsigned short)(r >> 16);
}

// Tensor Data Mover: 1-line 2D tile of `nelem` bf16 elements, global -> LDS.
// D# packing per CDNA5 ISA (count=1, type=2,
// data_size=1 (2 bytes), workgroup_mask=0: not in a cluster).
// This toolchain exposes the 6-arg builtin:
//   (u32x4 g0, i32x8 g1, i32x4 g2, i32x4 g3, i32x8, i32 cpol)
__device__ __forceinline__ void tdm_load_1d(unsigned lds_addr,
                                            const void* gsrc,
                                            unsigned nelem) {
    unsigned long long ga = (unsigned long long)(uintptr_t)gsrc;
    u32x4 g0;
    g0[0] = 1u;                                         // count = 1
    g0[1] = lds_addr;                                   // LDS byte address
    g0[2] = (unsigned)(ga & 0xffffffffu);               // global addr [31:0]
    g0[3] = (unsigned)((ga >> 32) & 0x1ffffffu)         // global addr [56:32]
            | (2u << 30);                               // type = 2 ("image")
    i32x8 g1;
    g1[0] = (int)(1u << 16);                            // data_size = 2 bytes
    g1[1] = (int)((nelem & 0xffffu) << 16);             // tensor_dim0 [15:0]
    g1[2] = (int)(((nelem >> 16) & 0xffffu)             // tensor_dim0 [31:16]
            | (1u << 16));                              // tensor_dim1 = 1
    g1[3] = (int)((nelem & 0xffffu) << 16);             // tile_dim0
    g1[4] = 1;                                          // tile_dim1 = 1
    g1[5] = (int)nelem;                                 // tensor_dim0_stride
    g1[6] = 0;
    g1[7] = 0;
    i32x4 z4 = {0, 0, 0, 0};                            // groups 2/3 unused (<=2D)
    i32x8 z8 = {0, 0, 0, 0, 0, 0, 0, 0};
    __builtin_amdgcn_tensor_load_to_lds(g0, g1, z4, z4, z8, 0);
}

// ------------------------- prep kernels ------------------------------------

// x: NCHW f32 -> NHWC bf16 (channels contiguous for im2col K-slabs)
__global__ __launch_bounds__(256) void k_convert_x(const float* __restrict__ x,
                                                   unsigned short* __restrict__ xn) {
    int i = blockIdx.x * 256 + threadIdx.x;
    if (i >= BATCH * HW * CIN) return;
    int c = i & 255;
    int p = (i >> 8) % HW;
    int b = (i >> 8) / HW;
    xn[i] = f2bf(x[((size_t)b * CIN + c) * HW + p]);
}

// Pack OIHW f32 weights into the WMMA B-fragment layout, bf16:
// element index = ((kc*NT + nt)*32 + lane)*16 + (part*8 + idx)
// lane = (n%16) + laneHi*16 ; kk = part*16 + laneHi*8 + idx ; k = kc*32+kk
// k = r9*256 + cin, r9 = ky*3+kx
__global__ __launch_bounds__(256) void k_pack_w(const float* __restrict__ w,
                                                unsigned short* __restrict__ out,
                                                int NT) {
    int i = blockIdx.x * 256 + threadIdx.x;
    int total = KC_STEPS * NT * 32 * 16;
    if (i >= total) return;
    int e    = i & 15;
    int lane = (i >> 4) & 31;
    int nt   = (i >> 9) % NT;
    int kc   = (i >> 9) / NT;
    int part = e >> 3, idx = e & 7;
    int kk   = part * 16 + ((lane >> 4) & 1) * 8 + idx;
    int k    = kc * 32 + kk;
    int n    = nt * 16 + (lane & 15);
    int cin  = k & 255;
    int r9   = k >> 8;
    int ky = r9 / 3, kx = r9 % 3;
    float v = w[(((size_t)n * CIN + cin) * 3 + ky) * 3 + kx];
    out[i] = f2bf(v);
}

// Heads: combined [cls(27) | box(36) | pad(1)] -> N = 64
__global__ __launch_bounds__(256) void k_pack_wh(const float* __restrict__ wc,
                                                 const float* __restrict__ wb,
                                                 unsigned short* __restrict__ out) {
    const int NT = 4;
    int i = blockIdx.x * 256 + threadIdx.x;
    int total = KC_STEPS * NT * 32 * 16;
    if (i >= total) return;
    int e    = i & 15;
    int lane = (i >> 4) & 31;
    int nt   = (i >> 9) % NT;
    int kc   = (i >> 9) / NT;
    int part = e >> 3, idx = e & 7;
    int kk   = part * 16 + ((lane >> 4) & 1) * 8 + idx;
    int k    = kc * 32 + kk;
    int n    = nt * 16 + (lane & 15);
    int cin  = k & 255;
    int r9   = k >> 8;
    int ky = r9 / 3, kx = r9 % 3;
    float v = 0.0f;
    if (n < 27)      v = wc[(((size_t)n * CIN + cin) * 3 + ky) * 3 + kx];
    else if (n < 63) v = wb[(((size_t)(n - 27) * CIN + cin) * 3 + ky) * 3 + kx];
    out[i] = f2bf(v);
}

// Fold conv-bias + eval BN into per-channel scale/shift:
// y = acc*inv + (bias*inv + beta - mean*inv), inv = gamma*rsqrt(var+eps)
__global__ __launch_bounds__(256) void k_bn_prep(const float* __restrict__ bias,
                                                 const float* __restrict__ g,
                                                 const float* __restrict__ be,
                                                 const float* __restrict__ m,
                                                 const float* __restrict__ v,
                                                 float* __restrict__ ss) {
    int i = threadIdx.x;
    if (i >= CIN) return;
    float inv = g[i] * rsqrtf(v[i] + 1e-5f);
    ss[i]       = inv;
    ss[CIN + i] = bias[i] * inv + be[i] - m[i] * inv;
}

__global__ __launch_bounds__(64) void k_head_prep(const float* __restrict__ bc,
                                                  const float* __restrict__ bb,
                                                  float* __restrict__ ss) {
    int i = threadIdx.x;   // 64 threads
    ss[i] = 1.0f;
    float sh = 0.0f;
    if (i < 27)      sh = bc[i];
    else if (i < 63) sh = bb[i - 27];
    ss[64 + i] = sh;
}

// ------------------------- WMMA implicit-GEMM conv -------------------------
// M-tile = 64 pixels per block; 8 waves: wave w -> msub = w&3, n-group = w>>2.
// Weights: TDM double-buffered into LDS (one 16KB slab per K-step); B frags
// then come from ds_read_b128. A slab staged via registers (im2col w/ zero pad).
template <int NT, bool RELU, bool OUTBF>
__global__ __launch_bounds__(256) void k_gemm_conv(
        const unsigned short* __restrict__ xin,   // NHWC bf16 [MTOT][256]
        const unsigned short* __restrict__ wpk,   // packed B fragments
        const float* __restrict__ ss,             // scale[NT*16] | shift[NT*16]
        unsigned short* __restrict__ obf,         // bf16 out [MTOT][NT*16]
        float* __restrict__ of) {                 // f32 out  [MTOT][NT*16]
    __shared__ __align__(16) unsigned short As[64 * 32];        // 4 KB A slab
    __shared__ __align__(16) unsigned short Bs[2][NT * 512];    // 2 x NT KB

    const int t    = threadIdx.x;
    const int wave = t >> 5;
    const int lane = t & 31;
    const int m0   = blockIdx.x * 64;
    const int msub = wave & 3;
    const int nt0  = (wave >> 2) * (NT / 2);
    const int hi   = (lane >> 4) & 1;
    const int mrow = msub * 16 + (lane & 15);
    const unsigned CHUNK = (unsigned)(NT * 512);   // bf16 elems per K-step

    v8f acc[NT / 2];
#pragma unroll
    for (int i = 0; i < NT / 2; ++i)
#pragma unroll
        for (int j = 0; j < 8; ++j) acc[i][j] = 0.0f;

    // A-staging mapping: 256 threads -> (row 0..63, 16-byte quarter 0..3)
    const int sr = t >> 2;
    const int sq = t & 3;
    const int gm = m0 + sr;
    const int sb = gm / HW;
    const int sp = gm % HW;
    const int sy = sp / GRID_HW;
    const int sx = sp % GRID_HW;

    // Prologue: DMA weight slab for kc=0 into buffer 0 (wave 0 only; TDM
    // ignores EXEC but must be issued once per workgroup).
    if (wave == 0) {
        tdm_load_1d((unsigned)(uintptr_t)&Bs[0][0], wpk, CHUNK);
    }

    for (int kc = 0; kc < KC_STEPS; ++kc) {
        const int r9 = kc >> 3;             // 0..8 spatial tap
        const int c0 = (kc & 7) * 32;       // channel base
        const int ky = r9 / 3, kx = r9 % 3;
        const int ys = sy + ky - 1;
        const int xs = sx + kx - 1;
        uint4 av = make_uint4(0u, 0u, 0u, 0u);
        if (ys >= 0 && ys < GRID_HW && xs >= 0 && xs < GRID_HW) {
            av = *(const uint4*)(xin +
                  ((size_t)(sb * HW + ys * GRID_HW + xs)) * CIN + c0 + sq * 8);
        }
        __syncthreads();     // prior readers of As / of Bs[(kc+1)&1] are done
        *(uint4*)(As + sr * 32 + sq * 8) = av;

        if (wave == 0) {
            if (kc + 1 < KC_STEPS) {
                // warm L2 + kick TDM for the next slab into the other buffer
                __builtin_prefetch(wpk + (size_t)(kc + 1) * CHUNK + lane * 64, 0, 1);
                tdm_load_1d((unsigned)(uintptr_t)&Bs[(kc + 1) & 1][0],
                            wpk + (size_t)(kc + 1) * CHUNK, CHUNK);
                __builtin_amdgcn_s_wait_tensorcnt(1);   // slab kc complete
            } else {
                __builtin_amdgcn_s_wait_tensorcnt(0);
            }
        }
        __syncthreads();     // As staged + Bs[kc&1] DMA complete

        // A fragment: documented 16-bit 16x32 striping -> two ds_read_b128
        union { uint4 u[2]; v16bf v; } af;
        af.u[0] = *(const uint4*)(As + mrow * 32 + hi * 8);
        af.u[1] = *(const uint4*)(As + mrow * 32 + 16 + hi * 8);

        // Batch all B fragments from LDS, then the WMMA chain.
        const unsigned short* bbase = &Bs[kc & 1][0];
        union { uint4 u[2]; v16bf v; } bfrag[NT / 2];
#pragma unroll
        for (int tt = 0; tt < NT / 2; ++tt) {
            const uint4* wsrc = (const uint4*)(bbase +
                  ((size_t)(nt0 + tt) * 32 + lane) * 16);
            bfrag[tt].u[0] = wsrc[0];
            bfrag[tt].u[1] = wsrc[1];
        }
#pragma unroll
        for (int tt = 0; tt < NT / 2; ++tt) {
            acc[tt] = __builtin_amdgcn_wmma_f32_16x16x32_bf16(
                false, af.v, false, bfrag[tt].v, (short)0, acc[tt], false, false);
        }
    }

    // Epilogue: scale/shift (+ReLU), C layout: lane%16 = n, VGPR r -> m = r + hi*8
#pragma unroll
    for (int tt = 0; tt < NT / 2; ++tt) {
        const int n  = (nt0 + tt) * 16 + (lane & 15);
        const float sc = ss[n];
        const float sh = ss[NT * 16 + n];
#pragma unroll
        for (int vr = 0; vr < 8; ++vr) {
            const int ml = msub * 16 + hi * 8 + vr;
            float v = acc[tt][vr] * sc + sh;
            if (RELU) v = fmaxf(v, 0.0f);
            const size_t o = (size_t)(m0 + ml) * (NT * 16) + n;
            if (OUTBF) obf[o] = f2bf(v);
            else       of[o]  = v;
        }
    }
}

// ------------------------- decode ------------------------------------------
__global__ __launch_bounds__(256) void k_decode(const float* __restrict__ head,
                                                const float* __restrict__ anchors,
                                                float* __restrict__ boxes,
                                                float* __restrict__ scores) {
    int i = blockIdx.x * 256 + threadIdx.x;
    if (i >= BATCH * NANCH) return;
    int a = i % NANCH;
    int b = i / NANCH;
    int p  = a / 9;
    int aa = a % 9;
    const float* h = head + ((size_t)(b * HW + p)) * 64;
    float s1 = h[aa * 3 + 1];
    float s2 = h[aa * 3 + 2];
    float tx = h[27 + aa * 4 + 0];
    float ty = h[27 + aa * 4 + 1];
    float tw = h[27 + aa * 4 + 2];
    float th = h[27 + aa * 4 + 3];
    const float* an = anchors + (size_t)a * 4;
    float aw = an[2] - an[0], ah = an[3] - an[1];
    float acx = (an[0] + an[2]) * 0.5f, acy = (an[1] + an[3]) * 0.5f;
    float pcx = tx * aw + acx, pcy = ty * ah + acy;
    float pw = __expf(tw) * aw, ph = __expf(th) * ah;
    float x0 = fminf(fmaxf(pcx - pw * 0.5f, 0.0f), 1.0f);
    float y0 = fminf(fmaxf(pcy - ph * 0.5f, 0.0f), 1.0f);
    float x1 = fminf(fmaxf(pcx + pw * 0.5f, 0.0f), 1.0f);
    float y1 = fminf(fmaxf(pcy + ph * 0.5f, 0.0f), 1.0f);
    float* ob = boxes + (size_t)i * 4;
    ob[0] = x0; ob[1] = y0; ob[2] = x1; ob[3] = y1;
    scores[((size_t)b * 2 + 0) * NANCH + a] = s1;
    scores[((size_t)b * 2 + 1) * NANCH + a] = s2;
}

// ------------------------- top-K (bitonic, stable ties) --------------------
__global__ __launch_bounds__(512) void k_topk(const float* __restrict__ scores,
                                              const float* __restrict__ boxes,
                                              float* __restrict__ cbox,
                                              float* __restrict__ cscore) {
    __shared__ float key[4096];
    __shared__ int   kid[4096];
    const int bc = blockIdx.x;       // b*2 + cls
    const int b  = bc >> 1;
    const float* s = scores + (size_t)bc * NANCH;
    const int t = threadIdx.x;
    for (int e = t; e < 4096; e += 512) {
        if (e < NANCH) { key[e] = s[e]; kid[e] = e; }
        else           { key[e] = -3.0e38f; kid[e] = e; }
    }
    __syncthreads();
    for (int k = 2; k <= 4096; k <<= 1) {
        for (int j = k >> 1; j > 0; j >>= 1) {
            for (int i = t; i < 4096; i += 512) {
                int ixj = i ^ j;
                if (ixj > i) {
                    float ka = key[i];   int ia = kid[i];
                    float kb = key[ixj]; int ib = kid[ixj];
                    bool aBeforeB = (ka > kb) || (ka == kb && ia < ib);
                    bool descDir  = ((i & k) == 0);
                    bool doSwap   = descDir ? (!aBeforeB) : aBeforeB;
                    if (doSwap) {
                        key[i] = kb; kid[i] = ib;
                        key[ixj] = ka; kid[ixj] = ia;
                    }
                }
            }
            __syncthreads();
        }
    }
    for (int e = t; e < KCAND; e += 512) {
        int src = kid[e];
        const float* bx = boxes + ((size_t)b * NANCH + src) * 4;
        float* ob = cbox + ((size_t)bc * KCAND + e) * 4;
        ob[0] = bx[0]; ob[1] = bx[1]; ob[2] = bx[2]; ob[3] = bx[3];
        cscore[(size_t)bc * KCAND + e] = key[e];
    }
}

// ------------------------- greedy NMS + outputs ----------------------------
// d_out layout (concatenated, all written as f32):
//   det_boxes [64][2][300][4] | det_scores [64][2][300] |
//   labels    [64][2][300]    | keep       [64][2][300]
__global__ __launch_bounds__(320) void k_nms(const float* __restrict__ cbox,
                                             const float* __restrict__ cscore,
                                             float* __restrict__ out) {
    __shared__ float bx[KCAND * 4];
    __shared__ float area[KCAND];
    __shared__ int   keep[KCAND];
    const int bc  = blockIdx.x;
    const int cls = bc & 1;
    const float sthr = 0.3f;
    const float ithr = (cls == 0) ? 0.5f : 0.45f;
    const int t = threadIdx.x;
    if (t < KCAND) {
        const float* p = cbox + ((size_t)bc * KCAND + t) * 4;
        bx[t * 4 + 0] = p[0]; bx[t * 4 + 1] = p[1];
        bx[t * 4 + 2] = p[2]; bx[t * 4 + 3] = p[3];
        area[t] = (p[2] - p[0]) * (p[3] - p[1]);
        keep[t] = (cscore[(size_t)bc * KCAND + t] > sthr) ? 1 : 0;
    }
    __syncthreads();
    for (int i = 0; i < KCAND - 1; ++i) {   // sequential greedy, matches lax.scan
        if (t < KCAND && t > i && keep[t] && keep[i]) {
            float lx = fmaxf(bx[i * 4 + 0], bx[t * 4 + 0]);
            float ly = fmaxf(bx[i * 4 + 1], bx[t * 4 + 1]);
            float rx = fminf(bx[i * 4 + 2], bx[t * 4 + 2]);
            float ry = fminf(bx[i * 4 + 3], bx[t * 4 + 3]);
            float w = fmaxf(rx - lx, 0.0f);
            float h = fmaxf(ry - ly, 0.0f);
            float inter = w * h;
            float iou = inter / (area[i] + area[t] - inter + 1e-9f);
            if (iou > ithr) keep[t] = 0;
        }
        __syncthreads();
    }
    if (t == 0) {                            // cap at MAX_DET
        int c = 0;
        for (int j = 0; j < KCAND; ++j) {
            if (keep[j]) { ++c; if (c > MAXDET) keep[j] = 0; }
        }
    }
    __syncthreads();
    float* dbox   = out;
    float* dscore = out + (size_t)BATCH * 2 * KCAND * 4;
    float* dlab   = dscore + (size_t)BATCH * 2 * KCAND;
    float* dkeep  = dlab + (size_t)BATCH * 2 * KCAND;
    if (t < KCAND) {
        const float* p = cbox + ((size_t)bc * KCAND + t) * 4;
        float* ob = dbox + ((size_t)bc * KCAND + t) * 4;
        ob[0] = p[0]; ob[1] = p[1]; ob[2] = p[2]; ob[3] = p[3];
        dscore[(size_t)bc * KCAND + t] = cscore[(size_t)bc * KCAND + t];
        dlab[(size_t)bc * KCAND + t]  = (float)(cls + 1);
        dkeep[(size_t)bc * KCAND + t] = keep[t] ? 1.0f : 0.0f;
    }
}

// ------------------------- launch ------------------------------------------
extern "C" void kernel_launch(void* const* d_in, const int* in_sizes, int n_in,
                              void* d_out, int out_size, void* d_ws, size_t ws_size,
                              hipStream_t stream) {
    (void)in_sizes; (void)n_in; (void)out_size; (void)ws_size;
    const float* x   = (const float*)d_in[0];
    const float* w1  = (const float*)d_in[1];
    const float* b1  = (const float*)d_in[2];
    const float* g1  = (const float*)d_in[3];
    const float* be1 = (const float*)d_in[4];
    const float* m1  = (const float*)d_in[5];
    const float* v1  = (const float*)d_in[6];
    const float* w2  = (const float*)d_in[7];
    const float* b2  = (const float*)d_in[8];
    const float* g2  = (const float*)d_in[9];
    const float* be2 = (const float*)d_in[10];
    const float* m2  = (const float*)d_in[11];
    const float* v2  = (const float*)d_in[12];
    const float* wc  = (const float*)d_in[13];
    const float* bc  = (const float*)d_in[14];
    const float* wb  = (const float*)d_in[15];
    const float* bb  = (const float*)d_in[16];
    const float* anchors = (const float*)d_in[17];

    char* ws = (char*)d_ws;
    // workspace layout (bytes)
    const size_t XN_OFF  = 0;                          // 13,107,200 (bf16 NHWC x, later f2)
    const size_t F1_OFF  = XN_OFF + 13107200;          // 13,107,200 (bf16 f1, later f32 head out)
    const size_t W1_OFF  = F1_OFF + 13107200;          //  1,179,648
    const size_t W2_OFF  = W1_OFF + 1179648;           //  1,179,648
    const size_t WH_OFF  = W2_OFF + 1179648;           //    294,912
    const size_t SS1_OFF = WH_OFF + 294912;            //      2,048
    const size_t SS2_OFF = SS1_OFF + 2048;             //      2,048
    const size_t SH_OFF  = SS2_OFF + 2048;             //        512
    const size_t BOX_OFF = SH_OFF + 512;               //  3,686,400
    const size_t SCR_OFF = BOX_OFF + 3686400;          //  1,843,200
    const size_t CBX_OFF = SCR_OFF + 1843200;          //    614,400
    const size_t CSC_OFF = CBX_OFF + 614400;           //    153,600

    unsigned short* xn  = (unsigned short*)(ws + XN_OFF);
    unsigned short* f1  = (unsigned short*)(ws + F1_OFF);
    unsigned short* f2  = (unsigned short*)(ws + XN_OFF);   // reuse: x dead after conv1
    unsigned short* w1p = (unsigned short*)(ws + W1_OFF);
    unsigned short* w2p = (unsigned short*)(ws + W2_OFF);
    unsigned short* whp = (unsigned short*)(ws + WH_OFF);
    float* ss1   = (float*)(ws + SS1_OFF);
    float* ss2   = (float*)(ws + SS2_OFF);
    float* ssh   = (float*)(ws + SH_OFF);
    float* hout  = (float*)(ws + F1_OFF);                   // reuse: f1 dead after conv2
    float* boxes = (float*)(ws + BOX_OFF);
    float* scrs  = (float*)(ws + SCR_OFF);
    float* cbox  = (float*)(ws + CBX_OFF);
    float* cscr  = (float*)(ws + CSC_OFF);

    // 1) prep
    k_convert_x<<<(BATCH * HW * CIN + 255) / 256, 256, 0, stream>>>(x, xn);
    k_pack_w<<<(KC_STEPS * 16 * 512 + 255) / 256, 256, 0, stream>>>(w1, w1p, 16);
    k_pack_w<<<(KC_STEPS * 16 * 512 + 255) / 256, 256, 0, stream>>>(w2, w2p, 16);
    k_pack_wh<<<(KC_STEPS * 4 * 512 + 255) / 256, 256, 0, stream>>>(wc, wb, whp);
    k_bn_prep<<<1, 256, 0, stream>>>(b1, g1, be1, m1, v1, ss1);
    k_bn_prep<<<1, 256, 0, stream>>>(b2, g2, be2, m2, v2, ss2);
    k_head_prep<<<1, 64, 0, stream>>>(bc, bb, ssh);

    // 2) conv1 / conv2 / heads (implicit GEMM, WMMA bf16 + TDM weight staging)
    k_gemm_conv<16, true, true><<<MTOT / 64, 256, 0, stream>>>(xn, w1p, ss1, f1, nullptr);
    k_gemm_conv<16, true, true><<<MTOT / 64, 256, 0, stream>>>(f1, w2p, ss2, f2, nullptr);
    k_gemm_conv<4, false, false><<<MTOT / 64, 256, 0, stream>>>(f2, whp, ssh, nullptr, hout);

    // 3) decode + top-K + NMS
    k_decode<<<(BATCH * NANCH + 255) / 256, 256, 0, stream>>>(hout, anchors, boxes, scrs);
    k_topk<<<BATCH * 2, 512, 0, stream>>>(scrs, boxes, cbox, cscr);
    k_nms<<<BATCH * 2, 320, 0, stream>>>(cbox, cscr, (float*)d_out);
}
